// Attention_11003706212411
// MI455X (gfx1250) — compile-verified, matrix operand order checked by reading, not run on hardware
//
#include <hip/hip_runtime.h>
#include <hip/hip_bf16.h>

// Problem constants
#define B_    4
#define S_    2048
#define H_    768
#define NH_   12
#define NKV_  2
#define HD_   64
#define GRP_  (NH_ / NKV_)     // 6

typedef __attribute__((ext_vector_type(16))) __bf16 v16bf;
typedef __attribute__((ext_vector_type(8)))  float  v8f;

#define WMMA_BF16(A, Bm, C) \
  __builtin_amdgcn_wmma_f32_16x16x32_bf16(false, (A), false, (Bm), (short)0, (C), false, false)

union FragU { uint4 q[2]; v16bf v; };

// Load one 16x32 bf16 WMMA operand fragment (A layout; B mirrored with
// "row" = column index). p = lane's row pointer at k-offset 0; kh = lane>>4.
// Chunks: [kh*8 .. +7] and [16+kh*8 .. +7] (16B each).
__device__ __forceinline__ v16bf ld_frag(const unsigned short* p, int kh) {
  FragU u;
  u.q[0] = *reinterpret_cast<const uint4*>(p + kh * 8);
  u.q[1] = *reinterpret_cast<const uint4*>(p + 16 + kh * 8);
  return u.v;
}

__device__ __forceinline__ unsigned short f2bf(float f) {
  unsigned int u = __float_as_uint(f);
  u = (u + 0x7FFFu + ((u >> 16) & 1u)) >> 16;  // round-to-nearest-even
  return (unsigned short)u;
}

// ---------------------------------------------------------------------------
// Kernel 0: fp32 -> bf16 conversion
// ---------------------------------------------------------------------------
__global__ void cvt_f32_bf16(const float* __restrict__ in,
                             unsigned short* __restrict__ out, int n) {
  int i = blockIdx.x * blockDim.x + threadIdx.x;
  if (i < n) out[i] = f2bf(in[i]);
}

// ---------------------------------------------------------------------------
// RoPE / V epilogue for one 16-row accumulator group of a head tile.
// ---------------------------------------------------------------------------
__device__ __forceinline__ void qkv_epilogue(
    int ht, int mbase, int nl, int kh,
    const v8f& a0, const v8f& a1, const v8f& a2, const v8f& a3,
    unsigned short* __restrict__ qO, unsigned short* __restrict__ kO,
    unsigned short* __restrict__ vT) {
  if (ht < NH_ + NKV_) {
    const float kf = 0.28782313662425593f;  // ln(10000)/32
    const float f0 = __expf(-(float)nl * kf);
    const float f1 = __expf(-(float)(nl + 16) * kf);
#pragma unroll
    for (int r = 0; r < 8; ++r) {
      int m = mbase + r + 8 * kh;
      int b = m >> 11;
      int s = m & (S_ - 1);
      float c0, s0, c1, s1;
      __sincosf((float)s * f0, &s0, &c0);
      __sincosf((float)s * f1, &s1, &c1);
      float x0 = a0[r], x1 = a1[r], x2 = a2[r], x3 = a3[r];
      float o0 = x0 * c0 - x2 * s0;   // cols  0..15
      float o1 = x1 * c1 - x3 * s1;   // cols 16..31
      float o2 = x2 * c0 + x0 * s0;   // cols 32..47
      float o3 = x3 * c1 + x1 * s1;   // cols 48..63
      unsigned short* dst;
      if (ht < NH_)
        dst = qO + ((size_t)(b * NH_ + ht) * S_ + s) * 64;
      else
        dst = kO + ((size_t)(b * NKV_ + (ht - NH_)) * S_ + s) * 64;
      dst[ 0 + nl] = f2bf(o0);
      dst[16 + nl] = f2bf(o1);
      dst[32 + nl] = f2bf(o2);
      dst[48 + nl] = f2bf(o3);
    }
  } else {
    const int hv = ht - (NH_ + NKV_);
#pragma unroll
    for (int r = 0; r < 8; ++r) {
      int m = mbase + r + 8 * kh;
      int b = m >> 11;
      int s = m & (S_ - 1);
      unsigned short* base = vT + (size_t)(b * NKV_ + hv) * 64 * S_;
      base[(size_t)( 0 + nl) * S_ + s] = f2bf(a0[r]);
      base[(size_t)(16 + nl) * S_ + s] = f2bf(a1[r]);
      base[(size_t)(32 + nl) * S_ + s] = f2bf(a2[r]);
      base[(size_t)(48 + nl) * S_ + s] = f2bf(a3[r]);
    }
  }
}

// ---------------------------------------------------------------------------
// Kernel 1: fused Q/KV projection GEMM + RoPE. One wave = 32(M) x 64(N)
// tile (N-tile == one head), register double-buffered over K.
//  head-tile 0..11 -> Q (RoPE) ; 12..13 -> K (RoPE) ; 14..15 -> V (transposed)
// ---------------------------------------------------------------------------
__global__ __launch_bounds__(128)
void qkv_rope_gemm(const unsigned short* __restrict__ hb,
                   const unsigned short* __restrict__ wq,
                   const unsigned short* __restrict__ wkv,
                   unsigned short* __restrict__ qO,
                   unsigned short* __restrict__ kO,
                   unsigned short* __restrict__ vT) {
  const int lane = threadIdx.x & 31;
  const int wv   = blockIdx.x * 4 + (threadIdx.x >> 5);
  const int mt   = wv >> 4;   // 0..255 (32-row M tiles)
  const int ht   = wv & 15;   // head tile
  const int nl   = lane & 15;
  const int kh   = lane >> 4;

  const unsigned short* wbase;
  int nbase;
  if (ht < NH_) { wbase = wq;  nbase = ht * 64; }
  else          { wbase = wkv; nbase = (ht - NH_) * 64; }

  const unsigned short* arow0 = hb + (size_t)(mt * 32 + nl) * H_;
  const unsigned short* arow1 = hb + (size_t)(mt * 32 + 16 + nl) * H_;
  const unsigned short* br0 = wbase + (size_t)(nbase +  0 + nl) * H_;
  const unsigned short* br1 = wbase + (size_t)(nbase + 16 + nl) * H_;
  const unsigned short* br2 = wbase + (size_t)(nbase + 32 + nl) * H_;
  const unsigned short* br3 = wbase + (size_t)(nbase + 48 + nl) * H_;

  v8f cA0 = {}, cA1 = {}, cA2 = {}, cA3 = {};
  v8f cB0 = {}, cB1 = {}, cB2 = {}, cB3 = {};

  // prologue loads (k0 = 0)
  v16bf a0 = ld_frag(arow0, kh), a1 = ld_frag(arow1, kh);
  v16bf b0 = ld_frag(br0, kh), b1 = ld_frag(br1, kh);
  v16bf b2 = ld_frag(br2, kh), b3 = ld_frag(br3, kh);

  for (int k0 = 0; k0 < H_; k0 += 32) {
    const int kn = (k0 + 32 < H_) ? (k0 + 32) : 0;  // clamped: straight-line
    __builtin_prefetch(arow0 + kn + 512, 0, 3);     // WGP-scope prefetch
    v16bf na0 = ld_frag(arow0 + kn, kh), na1 = ld_frag(arow1 + kn, kh);
    v16bf nb0 = ld_frag(br0 + kn, kh),   nb1 = ld_frag(br1 + kn, kh);
    v16bf nb2 = ld_frag(br2 + kn, kh),   nb3 = ld_frag(br3 + kn, kh);

    cA0 = WMMA_BF16(a0, b0, cA0);
    cA1 = WMMA_BF16(a0, b1, cA1);
    cA2 = WMMA_BF16(a0, b2, cA2);
    cA3 = WMMA_BF16(a0, b3, cA3);
    cB0 = WMMA_BF16(a1, b0, cB0);
    cB1 = WMMA_BF16(a1, b1, cB1);
    cB2 = WMMA_BF16(a1, b2, cB2);
    cB3 = WMMA_BF16(a1, b3, cB3);

    a0 = na0; a1 = na1; b0 = nb0; b1 = nb1; b2 = nb2; b3 = nb3;
  }

  qkv_epilogue(ht, mt * 32,      nl, kh, cA0, cA1, cA2, cA3, qO, kO, vT);
  qkv_epilogue(ht, mt * 32 + 16, nl, kh, cB0, cB1, cB2, cB3, qO, kO, vT);
}

// ---------------------------------------------------------------------------
// Kernel 2: causal flash attention. One wave per (b, h, 32-row q-tile):
// two 16-row groups share each K/V fragment set (halves K/V traffic).
// K fragments pipelined one key-block ahead; V fragments issued before the
// softmax VALU stretch. Softmax runs in the log2 domain (hardware exp is
// base-2), which is mathematically identical to exp-domain softmax.
// ---------------------------------------------------------------------------
__global__ __launch_bounds__(32)
void flash_attn(const unsigned short* __restrict__ qO,
                const unsigned short* __restrict__ kO,
                const unsigned short* __restrict__ vT,
                unsigned short* __restrict__ attn) {
  __shared__ unsigned short P[32 * 32];   // bf16 probs: C-layout -> A-layout bridge
  const int lane = threadIdx.x;
  const int nl = lane & 15;
  const int kh = lane >> 4;
  const int qt = blockIdx.x & 63;         // 64 q-tiles of 32 rows
  const int bh = blockIdx.x >> 6;         // 0..47
  const int b  = bh / NH_;
  const int h  = bh % NH_;
  const int kvh = h / GRP_;

  const unsigned short* Qh = qO + (size_t)bh * S_ * 64;
  const unsigned short* Kh = kO + (size_t)(b * NKV_ + kvh) * S_ * 64;
  const unsigned short* Vh = vT + (size_t)(b * NKV_ + kvh) * 64 * S_;

  v16bf aQ[2][2];
#pragma unroll
  for (int g = 0; g < 2; ++g) {
    aQ[g][0] = ld_frag(Qh + (size_t)(qt * 32 + g * 16 + nl) * 64 +  0, kh);
    aQ[g][1] = ld_frag(Qh + (size_t)(qt * 32 + g * 16 + nl) * 64 + 32, kh);
  }

  v8f o[2][4] = {};
  float mrow[2][8], lrow[2][8];
#pragma unroll
  for (int g = 0; g < 2; ++g)
#pragma unroll
    for (int r = 0; r < 8; ++r) { mrow[g][r] = -1e30f; lrow[g][r] = 0.0f; }

  const int   kend = qt * 32 + 32;               // exclusive causal key bound
  const float sl2e = 0.18033688011112042f;       // (1/sqrt(64)) * log2(e)

  // preload K fragments for kb = 0 : kf[keyhalf][kchunk]
  v16bf kf[2][2];
#pragma unroll
  for (int j = 0; j < 2; ++j) {
    kf[j][0] = ld_frag(Kh + (size_t)(j * 16 + nl) * 64 +  0, kh);
    kf[j][1] = ld_frag(Kh + (size_t)(j * 16 + nl) * 64 + 32, kh);
  }

  for (int kb = 0; kb < kend; kb += 32) {
    // V fragments for this block: issue early (consumed after softmax)
    v16bf bv[4];
#pragma unroll
    for (int t = 0; t < 4; ++t)
      bv[t] = ld_frag(Vh + (size_t)(t * 16 + nl) * S_ + kb, kh);

    // scores: 4 tiles (2 row groups x 2 key halves), 8 WMMAs
    v8f s[2][2] = {};
#pragma unroll
    for (int g = 0; g < 2; ++g)
#pragma unroll
      for (int j = 0; j < 2; ++j) {
        s[g][j] = WMMA_BF16(aQ[g][0], kf[j][0], s[g][j]);
        s[g][j] = WMMA_BF16(aQ[g][1], kf[j][1], s[g][j]);
      }

    // preload next block's K fragments (clamped; straight-line)
    const int kn = (kb + 32 < kend) ? (kb + 32) : kb;
#pragma unroll
    for (int j = 0; j < 2; ++j) {
      kf[j][0] = ld_frag(Kh + (size_t)(kn + j * 16 + nl) * 64 +  0, kh);
      kf[j][1] = ld_frag(Kh + (size_t)(kn + j * 16 + nl) * 64 + 32, kh);
    }

    // ---- scale + causal mask + online softmax (log2 domain) ----
#pragma unroll
    for (int g = 0; g < 2; ++g) {
#pragma unroll
      for (int r = 0; r < 8; ++r) {
        int row = g * 16 + r + 8 * kh;
        int qr  = qt * 32 + row;
        float v0 = s[g][0][r] * sl2e; if (kb + nl      > qr) v0 = -1e30f;
        float v1 = s[g][1][r] * sl2e; if (kb + 16 + nl > qr) v1 = -1e30f;
        float t = fmaxf(v0, v1);
#pragma unroll
        for (int off = 8; off >= 1; off >>= 1) t = fmaxf(t, __shfl_xor(t, off, 16));
        float nm = fmaxf(mrow[g][r], t);
        float a  = exp2f(mrow[g][r] - nm);
        float p0 = exp2f(v0 - nm);
        float p1 = exp2f(v1 - nm);
        float ps = p0 + p1;
#pragma unroll
        for (int off = 8; off >= 1; off >>= 1) ps += __shfl_xor(ps, off, 16);
        lrow[g][r] = lrow[g][r] * a + ps;
        mrow[g][r] = nm;
        P[row * 32 + nl]      = f2bf(p0);
        P[row * 32 + 16 + nl] = f2bf(p1);
#pragma unroll
        for (int t2 = 0; t2 < 4; ++t2) o[g][t2][r] *= a;
      }
    }
    // LDS in-order within wave; fence compiler + drain DS stores.
    asm volatile("s_wait_dscnt 0" ::: "memory");
    v16bf aP0 = ld_frag(&P[nl * 32], kh);
    v16bf aP1 = ld_frag(&P[(16 + nl) * 32], kh);
#pragma unroll
    for (int t = 0; t < 4; ++t) {
      o[0][t] = WMMA_BF16(aP0, bv[t], o[0][t]);
      o[1][t] = WMMA_BF16(aP1, bv[t], o[1][t]);
    }
    asm volatile("s_wait_dscnt 0" ::: "memory");
  }

  // ---- normalize + store bf16 to [B*S, H] ----
#pragma unroll
  for (int g = 0; g < 2; ++g)
#pragma unroll
    for (int r = 0; r < 8; ++r) {
      int row = g * 16 + r + 8 * kh;
      float inv = 1.0f / lrow[g][r];
      size_t dst = ((size_t)b * S_ + qt * 32 + row) * H_ + h * 64;
#pragma unroll
      for (int t = 0; t < 4; ++t)
        attn[dst + t * 16 + nl] = f2bf(o[g][t][r] * inv);
    }
}

// ---------------------------------------------------------------------------
// Kernel 3: output projection out = attn @ wo^T (fp32). 32x64 tile per wave,
// register double-buffered over K.
// ---------------------------------------------------------------------------
__global__ __launch_bounds__(128)
void out_proj(const unsigned short* __restrict__ attn,
              const unsigned short* __restrict__ wo,
              float* __restrict__ out) {
  const int lane = threadIdx.x & 31;
  const int wv = blockIdx.x * 4 + (threadIdx.x >> 5);
  const int mt = wv / 12;   // 0..255 (32-row tiles)
  const int nt = wv % 12;
  const int nl = lane & 15;
  const int kh = lane >> 4;

  const unsigned short* arow0 = attn + (size_t)(mt * 32 + nl) * H_;
  const unsigned short* arow1 = attn + (size_t)(mt * 32 + 16 + nl) * H_;
  const unsigned short* br0 = wo + (size_t)(nt * 64 +  0 + nl) * H_;
  const unsigned short* br1 = wo + (size_t)(nt * 64 + 16 + nl) * H_;
  const unsigned short* br2 = wo + (size_t)(nt * 64 + 32 + nl) * H_;
  const unsigned short* br3 = wo + (size_t)(nt * 64 + 48 + nl) * H_;

  v8f cA0 = {}, cA1 = {}, cA2 = {}, cA3 = {};
  v8f cB0 = {}, cB1 = {}, cB2 = {}, cB3 = {};

  v16bf a0 = ld_frag(arow0, kh), a1 = ld_frag(arow1, kh);
  v16bf b0 = ld_frag(br0, kh), b1 = ld_frag(br1, kh);
  v16bf b2 = ld_frag(br2, kh), b3 = ld_frag(br3, kh);

  for (int k0 = 0; k0 < H_; k0 += 32) {
    const int kn = (k0 + 32 < H_) ? (k0 + 32) : 0;
    __builtin_prefetch(arow0 + kn + 512, 0, 3);
    v16bf na0 = ld_frag(arow0 + kn, kh), na1 = ld_frag(arow1 + kn, kh);
    v16bf nb0 = ld_frag(br0 + kn, kh),   nb1 = ld_frag(br1 + kn, kh);
    v16bf nb2 = ld_frag(br2 + kn, kh),   nb3 = ld_frag(br3 + kn, kh);

    cA0 = WMMA_BF16(a0, b0, cA0);
    cA1 = WMMA_BF16(a0, b1, cA1);
    cA2 = WMMA_BF16(a0, b2, cA2);
    cA3 = WMMA_BF16(a0, b3, cA3);
    cB0 = WMMA_BF16(a1, b0, cB0);
    cB1 = WMMA_BF16(a1, b1, cB1);
    cB2 = WMMA_BF16(a1, b2, cB2);
    cB3 = WMMA_BF16(a1, b3, cB3);

    a0 = na0; a1 = na1; b0 = nb0; b1 = nb1; b2 = nb2; b3 = nb3;
  }

#pragma unroll
  for (int r = 0; r < 8; ++r) {
    size_t mA = (size_t)(mt * 32 + r + 8 * kh);
    size_t mB = mA + 16;
    out[mA * H_ + nt * 64 +  0 + nl] = cA0[r];
    out[mA * H_ + nt * 64 + 16 + nl] = cA1[r];
    out[mA * H_ + nt * 64 + 32 + nl] = cA2[r];
    out[mA * H_ + nt * 64 + 48 + nl] = cA3[r];
    out[mB * H_ + nt * 64 +  0 + nl] = cB0[r];
    out[mB * H_ + nt * 64 + 16 + nl] = cB1[r];
    out[mB * H_ + nt * 64 + 32 + nl] = cB2[r];
    out[mB * H_ + nt * 64 + 48 + nl] = cB3[r];
  }
}

// ---------------------------------------------------------------------------
extern "C" void kernel_launch(void* const* d_in, const int* in_sizes, int n_in,
                              void* d_out, int out_size, void* d_ws, size_t ws_size,
                              hipStream_t stream) {
  const float* hs  = (const float*)d_in[0];
  const float* wq  = (const float*)d_in[1];
  const float* wkv = (const float*)d_in[2];
  const float* wo  = (const float*)d_in[3];
  float* out = (float*)d_out;

  unsigned char* ws = (unsigned char*)d_ws;
  size_t off = 0;
  unsigned short* hb    = (unsigned short*)(ws + off); off += (size_t)B_ * S_ * H_ * 2;
  unsigned short* wqb   = (unsigned short*)(ws + off); off += (size_t)H_ * H_ * 2;
  unsigned short* wkvb  = (unsigned short*)(ws + off); off += (size_t)2 * NKV_ * HD_ * H_ * 2;
  unsigned short* wob   = (unsigned short*)(ws + off); off += (size_t)H_ * H_ * 2;
  unsigned short* qb    = (unsigned short*)(ws + off); off += (size_t)B_ * NH_ * S_ * HD_ * 2;
  unsigned short* kb    = (unsigned short*)(ws + off); off += (size_t)B_ * NKV_ * S_ * HD_ * 2;
  unsigned short* vtb   = (unsigned short*)(ws + off); off += (size_t)B_ * NKV_ * HD_ * S_ * 2;
  unsigned short* attnb = (unsigned short*)(ws + off); off += (size_t)B_ * S_ * H_ * 2;

  int n;
  n = B_ * S_ * H_;        cvt_f32_bf16<<<(n + 255) / 256, 256, 0, stream>>>(hs,  hb,   n);
  n = H_ * H_;             cvt_f32_bf16<<<(n + 255) / 256, 256, 0, stream>>>(wq,  wqb,  n);
  n = 2 * NKV_ * HD_ * H_; cvt_f32_bf16<<<(n + 255) / 256, 256, 0, stream>>>(wkv, wkvb, n);
  n = H_ * H_;             cvt_f32_bf16<<<(n + 255) / 256, 256, 0, stream>>>(wo,  wob,  n);

  // 256 M-tiles (32 rows) x 16 head-tiles = 4096 waves, 4 waves/block
  qkv_rope_gemm<<<1024, 128, 0, stream>>>(hb, wqb, wkvb, qb, kb, vtb);
  // 48 (b,h) pairs x 64 q-tiles (32 rows), 1 wave/block
  flash_attn<<<48 * 64, 32, 0, stream>>>(qb, kb, vtb, attnb);
  // 256 M-tiles x 12 N-tiles = 3072 waves, 4 waves/block
  out_proj<<<768, 128, 0, stream>>>(attnb, wob, out);
}
